// HybridTransformerBlock_4501125726446
// MI455X (gfx1250) — compile-verified, matrix operand order checked by reading, not run on hardware
//
#include <hip/hip_runtime.h>
#include <hip/hip_bf16.h>

// ---- problem constants (match reference) ----
#define Bq   2
#define Sq   4096
#define Dq   2048
#define Hq   16
#define HDq  128
#define FFq  8192
#define Wq_  256
#define NBq  16            // S / W
#define Mq   (Bq * Sq)     // 8192 rows
#define EPSq 1e-6f

typedef __bf16 bf16;
typedef __attribute__((ext_vector_type(16))) __bf16 v16bf;
typedef __attribute__((ext_vector_type(8)))  float  v8f;

// =====================================================================
// Row RMSNorm: x[row][0..Dn) fp32 -> out bf16, weight fp32
// =====================================================================
__global__ __launch_bounds__(256) void k_rmsnorm_row(const float* __restrict__ x,
                                                     const float* __restrict__ wgt,
                                                     bf16* __restrict__ out, int Dn) {
    int row = blockIdx.x;
    int tid = threadIdx.x;
    const float* xr = x + (size_t)row * Dn;
    float ss = 0.f;
    for (int i = tid; i < Dn; i += 256) { float v = xr[i]; ss += v * v; }
    __shared__ float red[256];
    red[tid] = ss; __syncthreads();
    for (int s = 128; s > 0; s >>= 1) {
        if (tid < s) red[tid] += red[tid + s];
        __syncthreads();
    }
    float inv = rsqrtf(red[0] / (float)Dn + EPSq);
    bf16* orow = out + (size_t)row * Dn;
    for (int i = tid; i < Dn; i += 256) orow[i] = (bf16)(xr[i] * inv * wgt[i]);
}

// =====================================================================
// Weight convert + transpose: Wt[n][k] = (bf16) W[k][n]
// tile 32(k) x 256(n), 256 threads; coalesced f32 reads, 64B bf16 stores
// =====================================================================
__global__ __launch_bounds__(256) void k_wt_bf16(const float* __restrict__ W,
                                                 bf16* __restrict__ Wt, int K, int N) {
    __shared__ bf16 tile[32][256];
    const int bk = blockIdx.x, bn = blockIdx.y;
    const int t = threadIdx.x;
    {
        int kr = t >> 3, nc = (t & 7) * 32;
        const float* src = W + (size_t)(bk * 32 + kr) * N + bn * 256 + nc;
        #pragma unroll
        for (int i = 0; i < 32; i++) tile[kr][nc + i] = (bf16)src[i];
    }
    __syncthreads();
    {
        int n = t;
        v16bf v0, v1;
        #pragma unroll
        for (int k = 0; k < 16; k++) { v0[k] = tile[k][n]; v1[k] = tile[k + 16][n]; }
        bf16* dst = Wt + (size_t)(bn * 256 + n) * K + bk * 32;
        *(v16bf*)&dst[0]  = v0;
        *(v16bf*)&dst[16] = v1;
    }
}

// =====================================================================
// bf16 GEMM: C[M,N] = A_bf16[M,K] @ Wt_bf16[N,K]^T   (Wt is n-major)
//   EPI 0: store bf16 C         EPI 1: store f32 C + resid
// WG: 256 thr = 8 waves (4 along M x 2 along N); WG tile 256x128;
// wave tile 64x64 (16 WMMA / K-step); double-buffered LDS; K-step 32.
// =====================================================================
template <int EPI>
__global__ __launch_bounds__(256) void k_gemm_bf16(const bf16* __restrict__ A,
                                                   const bf16* __restrict__ Bt,
                                                   void* __restrict__ Cout,
                                                   const float* __restrict__ resid,
                                                   int M, int N, int K) {
    __shared__ alignas(32) bf16 sA[2][256 * 32];   // 16 KB x2
    __shared__ alignas(32) bf16 sB[2][128 * 32];   //  8 KB x2

    const int bm = blockIdx.x, bn = blockIdx.y;
    const int tid = threadIdx.x;
    const int wv = tid >> 5;
    const int lane = tid & 31;
    const int wm = wv & 3;          // 4 waves along M
    const int wn = wv >> 2;         // 2 waves along N
    const int lhi = lane >> 4;
    const int llo = lane & 15;

    v8f acc[4][4];
    #pragma unroll
    for (int mt = 0; mt < 4; mt++)
        #pragma unroll
        for (int nt = 0; nt < 4; nt++) acc[mt][nt] = (v8f)(0.f);

    // staging pointers (per thread)
    const bf16* aptr = A  + (size_t)(bm * 256 + tid) * K;             // row tid, 64B
    const int   brow = tid >> 1, bcol = (tid & 1) * 16;
    const bf16* bptr = Bt + (size_t)(bn * 128 + brow) * K + bcol;     // 32B

    const int nk = K >> 5;
    v16bf av0, av1, bv;

    // prologue: tile 0
    av0 = *(const v16bf*)(aptr + 0);
    av1 = *(const v16bf*)(aptr + 16);
    bv  = *(const v16bf*)(bptr + 0);
    *(v16bf*)&sA[0][tid * 32]          = av0;
    *(v16bf*)&sA[0][tid * 32 + 16]     = av1;
    *(v16bf*)&sB[0][brow * 32 + bcol]  = bv;

    for (int i = 0; i < nk; i++) {
        __syncthreads();
        const int cur = i & 1;
        if (i + 1 < nk) {
            int k0 = (i + 1) << 5;
            av0 = *(const v16bf*)(aptr + k0);
            av1 = *(const v16bf*)(aptr + k0 + 16);
            bv  = *(const v16bf*)(bptr + k0);
            if (i + 2 < nk) {
                __builtin_prefetch(aptr + ((i + 2) << 5), 0, 1);
                __builtin_prefetch(bptr + ((i + 2) << 5), 0, 1);
            }
        }
        // fragments + 16 WMMAs
        {
            const int kofs = lhi * 16;
            v16bf af[4], bfr[4];
            #pragma unroll
            for (int mt = 0; mt < 4; mt++)
                af[mt] = *(const v16bf*)&sA[cur][(wm * 64 + mt * 16 + llo) * 32 + kofs];
            #pragma unroll
            for (int nt = 0; nt < 4; nt++)
                bfr[nt] = *(const v16bf*)&sB[cur][(wn * 64 + nt * 16 + llo) * 32 + kofs];
            #pragma unroll
            for (int mt = 0; mt < 4; mt++)
                #pragma unroll
                for (int nt = 0; nt < 4; nt++)
                    acc[mt][nt] = __builtin_amdgcn_wmma_f32_16x16x32_bf16(
                        false, af[mt], false, bfr[nt], (short)0, acc[mt][nt], false, false);
        }
        if (i + 1 < nk) {
            const int nxt = cur ^ 1;
            *(v16bf*)&sA[nxt][tid * 32]         = av0;
            *(v16bf*)&sA[nxt][tid * 32 + 16]    = av1;
            *(v16bf*)&sB[nxt][brow * 32 + bcol] = bv;
        }
    }

    // epilogue (C layout: lane%16 = N col, VGPR v + 8*(lane>=16) = M row)
    const int rsel = lhi * 8;
    #pragma unroll
    for (int mt = 0; mt < 4; mt++) {
        int row0 = bm * 256 + wm * 64 + mt * 16 + rsel;
        #pragma unroll
        for (int nt = 0; nt < 4; nt++) {
            int col = bn * 128 + wn * 64 + nt * 16 + llo;
            #pragma unroll
            for (int v = 0; v < 8; v++) {
                size_t idx = (size_t)(row0 + v) * N + col;
                if constexpr (EPI == 0) {
                    ((bf16*)Cout)[idx] = (bf16)acc[mt][nt][v];
                } else {
                    ((float*)Cout)[idx] = acc[mt][nt][v] + resid[idx];
                }
            }
        }
    }
}

// =====================================================================
// Per-head RMSNorm + RoPE for q,k; transpose q,k,v to [B,H,S,HD] bf16
// =====================================================================
__global__ __launch_bounds__(128) void k_qkv_rope(const bf16* __restrict__ qb,
                                                  const bf16* __restrict__ kb,
                                                  const bf16* __restrict__ vb,
                                                  const float* __restrict__ qn_w,
                                                  const float* __restrict__ kn_w,
                                                  bf16* __restrict__ qr,
                                                  bf16* __restrict__ kr,
                                                  bf16* __restrict__ vt) {
    int id = blockIdx.x;                    // b*S*H + s*H + h
    int h = id % Hq;
    int s = (id / Hq) % Sq;
    int b = id / (Hq * Sq);
    int d = threadIdx.x;

    size_t src = (((size_t)b * Sq + s) * Hq + h) * HDq + d;
    size_t dst = (((size_t)b * Hq + h) * Sq + s) * HDq + d;

    float q = (float)qb[src];
    float k = (float)kb[src];

    __shared__ float rq[128], rk[128];
    __shared__ float qsh[128], ksh[128];
    rq[d] = q * q; rk[d] = k * k;
    __syncthreads();
    for (int st = 64; st > 0; st >>= 1) {
        if (d < st) { rq[d] += rq[d + st]; rk[d] += rk[d + st]; }
        __syncthreads();
    }
    float qn = q * rsqrtf(rq[0] / 128.f + EPSq) * qn_w[d];
    float kn = k * rsqrtf(rk[0] / 128.f + EPSq) * kn_w[d];
    qsh[d] = qn; ksh[d] = kn;
    __syncthreads();

    int i = d & 63;
    float inv = __powf(10000.0f, -((float)i) / 64.0f);
    float ang = (float)s * inv;
    float sn, cs;
    __sincosf(ang, &sn, &cs);
    float qo, ko;
    if (d < 64) {
        qo = qsh[d] * cs - qsh[d + 64] * sn;
        ko = ksh[d] * cs - ksh[d + 64] * sn;
    } else {
        qo = qsh[d - 64] * sn + qsh[d] * cs;
        ko = ksh[d - 64] * sn + ksh[d] * cs;
    }
    qr[dst] = (bf16)qo;
    kr[dst] = (bf16)ko;
    vt[dst] = vb[src];
}

// =====================================================================
// Sliding-window attention w/ sink, flash-style online softmax.
// grid = B*H*NB*2 ; WG = 256 thr (8 waves); each WG: 128 q rows x 512 keys
// =====================================================================
__global__ __launch_bounds__(256) void k_attn(const bf16* __restrict__ qr,
                                              const bf16* __restrict__ kr,
                                              const bf16* __restrict__ vt,
                                              const float* __restrict__ sink,
                                              bf16* __restrict__ aout) {
    __shared__ alignas(32) bf16 sK[32 * 128];       // [key][hd], 8 KB
    __shared__ alignas(32) bf16 sV[128 * 32];       // [hd][key] (transposed), 8 KB
    __shared__ alignas(32) bf16 pP[8][16][32];      // per-wave P tile, 8 KB

    const int id = blockIdx.x;
    const int half = id & 1;
    const int blk = (id >> 1) & (NBq - 1);
    const int h   = (id >> 5) & (Hq - 1);
    const int b   = id >> 9;

    const int tid = threadIdx.x;
    const int wv = tid >> 5;
    const int lane = tid & 31;
    const int llo = lane & 15;
    const int lhi = lane >> 4;

    const float SCALE = 0.08838834764831845f;       // 1/sqrt(128)
    const int qi0 = half * 128 + wv * 16;           // within-block query base
    const size_t head_base = ((size_t)b * Hq + h) * Sq;

    v16bf qf[4];
    {
        int srow = blk * Wq_ + qi0 + llo;
        const bf16* qp = qr + (head_base + srow) * HDq + lhi * 16;
        #pragma unroll
        for (int j = 0; j < 4; j++) qf[j] = *(const v16bf*)(qp + j * 32);
    }

    v8f outv[8];
    #pragma unroll
    for (int nt = 0; nt < 8; nt++) outv[nt] = (v8f)(0.f);
    float m[8], l[8];
    float sinkv = sink[h];
    #pragma unroll
    for (int v = 0; v < 8; v++) { m[v] = sinkv; l[v] = 1.0f; }

    const int kvbase = blk * Wq_ - Wq_;             // global pos of window key 0

    for (int c = 0; c < 16; c++) {                  // 16 chunks of 32 keys
        {
            int r = tid >> 3;
            int cb = (tid & 7) * 16;
            int kpos = kvbase + c * 32 + r;
            if (kpos >= 0) {
                size_t g = (head_base + kpos) * HDq + cb;
                *(v16bf*)&sK[r * 128 + cb] = *(const v16bf*)(kr + g);
                v16bf vvv = *(const v16bf*)(vt + g);
                #pragma unroll
                for (int i = 0; i < 16; i++) sV[(cb + i) * 32 + r] = vvv[i];
            } else {
                #pragma unroll
                for (int i = 0; i < 16; i++) {
                    sK[r * 128 + cb + i] = (bf16)0.f;
                    sV[(cb + i) * 32 + r] = (bf16)0.f;
                }
            }
        }
        __syncthreads();

        v8f s0 = (v8f)(0.f), s1 = (v8f)(0.f);
        #pragma unroll
        for (int j = 0; j < 4; j++) {
            int kk = j * 32 + lhi * 16;
            v16bf b0 = *(const v16bf*)&sK[(llo) * 128 + kk];
            v16bf b1 = *(const v16bf*)&sK[(16 + llo) * 128 + kk];
            s0 = __builtin_amdgcn_wmma_f32_16x16x32_bf16(false, qf[j], false, b0,
                                                         (short)0, s0, false, false);
            s1 = __builtin_amdgcn_wmma_f32_16x16x32_bf16(false, qf[j], false, b1,
                                                         (short)0, s1, false, false);
        }

        float p0[8], p1[8];
        const int kj0 = c * 32 + llo;
        const int kj1 = kj0 + 16;
        const int rsel = lhi * 8;
        #pragma unroll
        for (int v = 0; v < 8; v++) {
            int qi = qi0 + rsel + v;
            bool ok0 = (kj0 > qi) && (kj0 <= qi + Wq_) && (blk > 0 || kj0 >= Wq_);
            bool ok1 = (kj1 > qi) && (kj1 <= qi + Wq_) && (blk > 0 || kj1 >= Wq_);
            float a0 = ok0 ? s0[v] * SCALE : -1e30f;
            float a1 = ok1 ? s1[v] * SCALE : -1e30f;
            float rm = fmaxf(a0, a1);
            #pragma unroll
            for (int off = 1; off < 16; off <<= 1)
                rm = fmaxf(rm, __shfl_xor(rm, off, 32));
            float mn = fmaxf(m[v], rm);
            float f = __expf(m[v] - mn);
            p0[v] = __expf(a0 - mn);
            p1[v] = __expf(a1 - mn);
            float rs = p0[v] + p1[v];
            #pragma unroll
            for (int off = 1; off < 16; off <<= 1)
                rs += __shfl_xor(rs, off, 32);
            l[v] = l[v] * f + rs;
            m[v] = mn;
            #pragma unroll
            for (int nt = 0; nt < 8; nt++) outv[nt][v] *= f;
        }

        #pragma unroll
        for (int v = 0; v < 8; v++) {
            pP[wv][rsel + v][llo]      = (bf16)p0[v];
            pP[wv][rsel + v][16 + llo] = (bf16)p1[v];
        }
        v16bf pa = *(const v16bf*)&pP[wv][llo][lhi * 16];

        #pragma unroll
        for (int nt = 0; nt < 8; nt++) {
            v16bf bvv = *(const v16bf*)&sV[(nt * 16 + llo) * 32 + lhi * 16];
            outv[nt] = __builtin_amdgcn_wmma_f32_16x16x32_bf16(false, pa, false, bvv,
                                                               (short)0, outv[nt], false, false);
        }
        __syncthreads();
    }

    const int rsel = lhi * 8;
    #pragma unroll
    for (int nt = 0; nt < 8; nt++) {
        int col = h * HDq + nt * 16 + llo;
        #pragma unroll
        for (int v = 0; v < 8; v++) {
            int srow = blk * Wq_ + qi0 + rsel + v;
            float o = outv[nt][v] / l[v];
            aout[((size_t)b * Sq + srow) * Dq + col] = (bf16)o;
        }
    }
}

// =====================================================================
// g = silu(g) * u  (in-place on g), bf16
// =====================================================================
__global__ __launch_bounds__(256) void k_silu_mul(bf16* __restrict__ g,
                                                  const bf16* __restrict__ u, size_t n) {
    size_t stride = (size_t)gridDim.x * blockDim.x;
    for (size_t i = (size_t)blockIdx.x * blockDim.x + threadIdx.x; i < n; i += stride) {
        float gv = (float)g[i];
        float uv = (float)u[i];
        float s = gv / (1.0f + __expf(-gv));
        g[i] = (bf16)(s * uv);
    }
}

// =====================================================================
extern "C" void kernel_launch(void* const* d_in, const int* in_sizes, int n_in,
                              void* d_out, int out_size, void* d_ws, size_t ws_size,
                              hipStream_t stream) {
    const float* x      = (const float*)d_in[0];
    const float* Wqw    = (const float*)d_in[1];
    const float* Wkw    = (const float*)d_in[2];
    const float* Wvw    = (const float*)d_in[3];
    const float* Wow    = (const float*)d_in[4];
    const float* qn_w   = (const float*)d_in[5];
    const float* kn_w   = (const float*)d_in[6];
    const float* sinkl  = (const float*)d_in[7];
    const float* anorm  = (const float*)d_in[8];
    const float* fnorm  = (const float*)d_in[9];
    const float* Wgw    = (const float*)d_in[10];
    const float* Wuw    = (const float*)d_in[11];
    const float* Wdw    = (const float*)d_in[12];

    char* base = (char*)d_ws;
    size_t off = 0;
    auto alloc = [&](size_t bytes) -> void* {
        void* r = base + off;
        off += (bytes + 255) & ~(size_t)255;
        return r;
    };
    const size_t MD2 = (size_t)Mq * Dq * sizeof(bf16);
    const size_t DD2 = (size_t)Dq * Dq * sizeof(bf16);
    const size_t DF2 = (size_t)Dq * FFq * sizeof(bf16);
    bf16* hb   = (bf16*)alloc(MD2);                  // normed input / reused as h2
    bf16* qb   = (bf16*)alloc(MD2);                  // q proj / reused as attn_out
    bf16* kb   = (bf16*)alloc(MD2);
    bf16* vb   = (bf16*)alloc(MD2);
    bf16* qrr  = (bf16*)alloc(MD2);
    bf16* krr  = (bf16*)alloc(MD2);
    bf16* vtt  = (bf16*)alloc(MD2);
    float* x1  = (float*)alloc((size_t)Mq * Dq * sizeof(float));
    bf16* gbuf = (bf16*)alloc((size_t)Mq * FFq * sizeof(bf16));
    bf16* ubuf = (bf16*)alloc((size_t)Mq * FFq * sizeof(bf16));
    bf16* Wqt  = (bf16*)alloc(DD2);
    bf16* Wkt  = (bf16*)alloc(DD2);
    bf16* Wvt  = (bf16*)alloc(DD2);
    bf16* Wot  = (bf16*)alloc(DD2);
    bf16* Wgt  = (bf16*)alloc(DF2);
    bf16* Wut  = (bf16*)alloc(DF2);
    bf16* Wdt  = (bf16*)alloc(DF2);
    (void)ws_size; (void)in_sizes; (void)n_in; (void)out_size;

    dim3 blk256(256), blk128(128);
    dim3 gD(Mq / 256, Dq / 128);     // (32,16)
    dim3 gF(Mq / 256, FFq / 128);    // (32,64)
    dim3 tDD(Dq / 32, Dq / 256);     // (64,8)
    dim3 tDF(Dq / 32, FFq / 256);    // (64,32)
    dim3 tFD(FFq / 32, Dq / 256);    // (256,8)

    // 0) one-time weight convert + transpose (fp32 [K][N] -> bf16 [N][K])
    k_wt_bf16<<<tDD, blk256, 0, stream>>>(Wqw, Wqt, Dq, Dq);
    k_wt_bf16<<<tDD, blk256, 0, stream>>>(Wkw, Wkt, Dq, Dq);
    k_wt_bf16<<<tDD, blk256, 0, stream>>>(Wvw, Wvt, Dq, Dq);
    k_wt_bf16<<<tDD, blk256, 0, stream>>>(Wow, Wot, Dq, Dq);
    k_wt_bf16<<<tDF, blk256, 0, stream>>>(Wgw, Wgt, Dq, FFq);
    k_wt_bf16<<<tDF, blk256, 0, stream>>>(Wuw, Wut, Dq, FFq);
    k_wt_bf16<<<tFD, blk256, 0, stream>>>(Wdw, Wdt, FFq, Dq);

    // 1) attn pre-norm
    k_rmsnorm_row<<<Mq, blk256, 0, stream>>>(x, anorm, hb, Dq);
    // 2) QKV projections
    k_gemm_bf16<0><<<gD, blk256, 0, stream>>>(hb, Wqt, qb, nullptr, Mq, Dq, Dq);
    k_gemm_bf16<0><<<gD, blk256, 0, stream>>>(hb, Wkt, kb, nullptr, Mq, Dq, Dq);
    k_gemm_bf16<0><<<gD, blk256, 0, stream>>>(hb, Wvt, vb, nullptr, Mq, Dq, Dq);
    // 3) head norm + rope + transpose
    k_qkv_rope<<<Bq * Sq * Hq, blk128, 0, stream>>>(qb, kb, vb, qn_w, kn_w, qrr, krr, vtt);
    // 4) sliding-window attention (attn_out reuses qb)
    bf16* attn_out = qb;
    k_attn<<<Bq * Hq * NBq * 2, blk256, 0, stream>>>(qrr, krr, vtt, sinkl, attn_out);
    // 5) output proj + residual -> x1
    k_gemm_bf16<1><<<gD, blk256, 0, stream>>>(attn_out, Wot, x1, x, Mq, Dq, Dq);
    // 6) ffn pre-norm (h2 reuses hb)
    k_rmsnorm_row<<<Mq, blk256, 0, stream>>>(x1, fnorm, hb, Dq);
    // 7) gate / up projections
    k_gemm_bf16<0><<<gF, blk256, 0, stream>>>(hb, Wgt, gbuf, nullptr, Mq, FFq, Dq);
    k_gemm_bf16<0><<<gF, blk256, 0, stream>>>(hb, Wut, ubuf, nullptr, Mq, FFq, Dq);
    // 8) silu * up
    k_silu_mul<<<8192, blk256, 0, stream>>>(gbuf, ubuf, (size_t)Mq * FFq);
    // 9) down proj + residual -> d_out (fp32)
    k_gemm_bf16<1><<<gD, blk256, 0, stream>>>(gbuf, Wdt, d_out, x1, Mq, Dq, FFq);
}
